// nchwBRA_66812511256599
// MI455X (gfx1250) — matrix operand order repaired and checked by reading, not compiled, not run on hardware
//
#include <hip/hip_runtime.h>
#include <hip/hip_bf16.h>

// ---------------------------------------------------------------------------
// BiFormer BRA pipeline for MI455X (gfx1250, wave32, WMMA 16x16x32 f16)
// N=8, C=256, H=W=56, heads=8, hd=32, 7x7 regions of 8x8 tokens, TOPK=4
// ---------------------------------------------------------------------------

#define NB   8
#define CC   256
#define HH   56
#define WW   56
#define HWP  (HH * WW)            // 3136
#define NH   8
#define HD   32
#define NWIN 7
#define NR   49                   // regions
#define RT   64                   // tokens per region
#define TOPK 4
#define NKEY (TOPK * RT)          // 256 gathered keys

typedef __attribute__((ext_vector_type(16))) _Float16 v16h;
typedef __attribute__((ext_vector_type(8)))  float    v8f;

static __device__ inline v8f wmma_f16(v16h a, v16h b, v8f c) {
    return __builtin_amdgcn_wmma_f32_16x16x32_f16(false, a, false, b, (short)0, c,
                                                  false, false);
}

// A tile 16x32 (rows m0.., K k0..) from row-major f16 [*, lda]
static __device__ inline v16h load_a16(const _Float16* __restrict__ A, int lda,
                                       int m0, int k0, int lane) {
    int half = lane >> 4, r = lane & 15;
    const _Float16* p = A + (size_t)(m0 + r) * lda + k0;
    v16h a;
#pragma unroll
    for (int i = 0; i < 8; ++i) a[i] = p[half * 16 + i];
#pragma unroll
    for (int i = 0; i < 8; ++i) a[8 + i] = p[half * 16 + 8 + i];
    return a;
}

// B tile 32x16 (K k0.., cols n0..) from transposed f16 layout: element(k,c) at B[c*ldb + k]
static __device__ inline v16h load_bT16(const _Float16* __restrict__ B, int ldb,
                                        int k0, int n0, int lane) {
    int half = lane >> 4, c = lane & 15;
    const _Float16* p = B + (size_t)(n0 + c) * ldb + k0 + half * 16;
    v16h b;
#pragma unroll
    for (int i = 0; i < 16; ++i) b[i] = p[i];
    return b;
}

// ---------------------------------------------------------------------------
// Packing kernels
// ---------------------------------------------------------------------------
__global__ void pack_x_kernel(const float* __restrict__ x, _Float16* __restrict__ xT) {
    size_t t = (size_t)blockIdx.x * blockDim.x + threadIdx.x;
    if (t >= (size_t)NB * HWP * CC) return;
    int c = (int)(t & (CC - 1));
    size_t rest = t >> 8;
    int p = (int)(rest % HWP);
    int n = (int)(rest / HWP);
    xT[t] = (_Float16)x[((size_t)n * CC + c) * HWP + p];
}

__global__ void cvt_f16_kernel(const float* __restrict__ src, _Float16* __restrict__ dst,
                               int count) {
    int t = blockIdx.x * blockDim.x + threadIdx.x;
    if (t < count) dst[t] = (_Float16)src[t];
}

// ---------------------------------------------------------------------------
// QKV projection GEMM epilogue scatter (one 16x16 D tile)
// ---------------------------------------------------------------------------
static __device__ inline void qkv_scatter(
    v8f acc, int n, int mbase, int pbase, int lane, const float* __restrict__ bias,
    _Float16* __restrict__ q16, _Float16* __restrict__ k16, _Float16* __restrict__ vT16,
    float* __restrict__ v32) {
    int half = lane >> 4, c = lane & 15;
    int p = pbase + c;
    int y = p / WW, x = p % WW;
    int reg = (y >> 3) * NWIN + (x >> 3);
    int tok = ((y & 7) << 3) + (x & 7);
#pragma unroll
    for (int j = 0; j < 8; ++j) {
        int o = mbase + half * 8 + j;
        float val = acc[j] + bias[o];
        int oc = o & (CC - 1);
        int which = o >> 8;               // 0=q, 1=k, 2=v
        int h = oc >> 5, hd = oc & 31;
        size_t base = (((size_t)(n * NH + h)) * NR + reg) * (RT * HD);
        if (which == 0) {
            q16[base + (size_t)tok * HD + hd] = (_Float16)val;
        } else if (which == 1) {
            k16[base + (size_t)tok * HD + hd] = (_Float16)val;
        } else {
            vT16[base + (size_t)hd * RT + tok] = (_Float16)val;
            v32[((size_t)n * CC + oc) * HWP + p] = val;
        }
    }
}

// ---------------------------------------------------------------------------
// QKV projection GEMM: qkv[o,p] = sum_c w[o,c]*x[n,c,p] + b[o]
// 2x2 register blocking: one wave computes a 32x32 D block (4 WMMA tiles),
// reusing each A/B fragment twice. Grid (98, 24, 8).
// ---------------------------------------------------------------------------
__global__ __launch_bounds__(32) void gemm_qkv_kernel(
    const _Float16* __restrict__ w16,   // [768][256]
    const _Float16* __restrict__ xT,    // [n][3136][256]
    const float*    __restrict__ bias,  // [768]
    _Float16* __restrict__ q16,         // [n][h][r][tok][hd]
    _Float16* __restrict__ k16,         // [n][h][r][tok][hd]
    _Float16* __restrict__ vT16,        // [n][h][r][hd][tok]
    float*    __restrict__ v32)         // [n][c][hw]
{
    int lane = threadIdx.x & 31;
    int n = blockIdx.z, m0 = blockIdx.y * 32, p0 = blockIdx.x * 32;
    const _Float16* B = xT + (size_t)n * HWP * CC;

    v8f acc00 = {}, acc01 = {}, acc10 = {}, acc11 = {};
#pragma unroll
    for (int k0 = 0; k0 < CC; k0 += 32) {
        v16h a0 = load_a16(w16, CC, m0, k0, lane);
        v16h a1 = load_a16(w16, CC, m0 + 16, k0, lane);
        v16h b0 = load_bT16(B, CC, k0, p0, lane);
        v16h b1 = load_bT16(B, CC, k0, p0 + 16, lane);
        acc00 = wmma_f16(a0, b0, acc00);
        acc01 = wmma_f16(a0, b1, acc01);
        acc10 = wmma_f16(a1, b0, acc10);
        acc11 = wmma_f16(a1, b1, acc11);
    }
    qkv_scatter(acc00, n, m0,      p0,      lane, bias, q16, k16, vT16, v32);
    qkv_scatter(acc01, n, m0,      p0 + 16, lane, bias, q16, k16, vT16, v32);
    qkv_scatter(acc10, n, m0 + 16, p0,      lane, bias, q16, k16, vT16, v32);
    qkv_scatter(acc11, n, m0 + 16, p0 + 16, lane, bias, q16, k16, vT16, v32);
}

// ---------------------------------------------------------------------------
// Region pooling: q_r/k_r[n][r][c] = mean over 64 tokens
// ---------------------------------------------------------------------------
__global__ void pool_kernel(const _Float16* __restrict__ q16,
                            const _Float16* __restrict__ k16,
                            float* __restrict__ q_r, float* __restrict__ k_r) {
    int t = blockIdx.x * blockDim.x + threadIdx.x;
    if (t >= NB * NR * CC) return;
    int c = t & (CC - 1);
    int rgn = (t >> 8) % NR;
    int n = t / (NR * CC);
    int h = c >> 5, hd = c & 31;
    size_t base = (((size_t)(n * NH + h)) * NR + rgn) * (RT * HD) + hd;
    float sq = 0.f, sk = 0.f;
#pragma unroll 4
    for (int tok = 0; tok < RT; ++tok) {
        sq += (float)q16[base + (size_t)tok * HD];
        sk += (float)k16[base + (size_t)tok * HD];
    }
    q_r[t] = sq * (1.0f / RT);
    k_r[t] = sk * (1.0f / RT);
}

// ---------------------------------------------------------------------------
// Routing: a_r = q_r @ k_r^T (49x49 per image), top-4 indices per query region
// ---------------------------------------------------------------------------
__global__ void route_kernel(const float* __restrict__ q_r, const float* __restrict__ k_r,
                             int* __restrict__ idx) {
    int t = blockIdx.x * blockDim.x + threadIdx.x;
    if (t >= NB * NR) return;
    int n = t / NR, i = t % NR;
    const float* q = q_r + ((size_t)n * NR + i) * CC;
    float s[NR];
    for (int j = 0; j < NR; ++j) {
        const float* k = k_r + ((size_t)n * NR + j) * CC;
        float d = 0.f;
#pragma unroll 8
        for (int c = 0; c < CC; ++c) d += q[c] * k[c];
        s[j] = d;
    }
#pragma unroll
    for (int sel = 0; sel < TOPK; ++sel) {
        float best = -3.4e38f;
        int bj = 0;
        for (int j = 0; j < NR; ++j)
            if (s[j] > best) { best = s[j]; bj = j; }
        idx[t * TOPK + sel] = bj;
        s[bj] = -3.4e38f;
    }
}

// ---------------------------------------------------------------------------
// Attention per (region, head, image): logits via WMMA -> softmax in LDS ->
// PV via WMMA. 128 threads = 4 waves; wave w owns tokens 16w..16w+15.
// LDS: 64 rows x 256 cols f32 = exactly 64 KB.
// ---------------------------------------------------------------------------
__global__ __launch_bounds__(128) void attn_kernel(
    const _Float16* __restrict__ q16, const _Float16* __restrict__ k16,
    const _Float16* __restrict__ vT16, const int* __restrict__ idx,
    float* __restrict__ attn32) {
    __shared__ float S[RT * NKEY];   // 64*256 f32

    int i_reg = blockIdx.x, h = blockIdx.y, n = blockIdx.z;
    int tid = threadIdx.x;
    int wave = tid >> 5, lane = tid & 31;
    int half = lane >> 4, r = lane & 15;

    const int* idx4 = idx + ((size_t)n * NR + i_reg) * TOPK;
    int regs[TOPK];
#pragma unroll
    for (int t = 0; t < TOPK; ++t) regs[t] = idx4[t];

    size_t hb = (size_t)(n * NH + h) * NR;   // head base (region units)
    int m0 = wave * 16;

    // Prefetch gathered k/v region blocks (data-dependent gather; warm L2/WGP$).
    // Each region block is 4 KB; 128 threads x 32 B strides cover it.
#pragma unroll
    for (int t = 0; t < TOPK; ++t) {
        const _Float16* kb = k16 + (hb + regs[t]) * (RT * HD);
        const _Float16* vb = vT16 + (hb + regs[t]) * (RT * HD);
        __builtin_prefetch(kb + (size_t)tid * 16, 0, 1);
        __builtin_prefetch(vb + (size_t)tid * 16, 0, 1);
    }

    // --- A tile: q rows m0..m0+15, K = hd (32) ---
    v16h aq;
    {
        const _Float16* p = q16 + (hb + i_reg) * (RT * HD) + (size_t)(m0 + r) * HD;
#pragma unroll
        for (int i = 0; i < 8; ++i) aq[i] = p[half * 16 + i];
#pragma unroll
        for (int i = 0; i < 8; ++i) aq[8 + i] = p[half * 16 + 8 + i];
    }

    // --- GEMM1: logits = scale * q @ kg^T ---
    for (int n0 = 0; n0 < NKEY; n0 += 16) {
        int reg2 = regs[n0 >> 6];
        const _Float16* kb = k16 + (hb + reg2) * (RT * HD);
        const _Float16* p = kb + (size_t)((n0 & 63) + r) * HD + half * 16;
        v16h bk;
#pragma unroll
        for (int i = 0; i < 16; ++i) bk[i] = p[i];
        v8f acc = {};
        acc = wmma_f16(aq, bk, acc);
#pragma unroll
        for (int j = 0; j < 8; ++j)
            S[(size_t)(m0 + half * 8 + j) * NKEY + n0 + r] = acc[j] * 0.0625f; // 256^-0.5
    }
    __syncthreads();

    // --- softmax over 256 keys; thread t handles row t (lane rotation vs banks) ---
    if (tid < RT) {
        float* row = S + (size_t)tid * NKEY;
        float mx = -3.4e38f;
        for (int j = 0; j < NKEY; ++j) {
            float v = row[(j + tid) & (NKEY - 1)];
            mx = fmaxf(mx, v);
        }
        float sum = 0.f;
        for (int j = 0; j < NKEY; ++j) {
            int jj = (j + tid) & (NKEY - 1);
            float e = __expf(row[jj] - mx);
            row[jj] = e;
            sum += e;
        }
        float inv = 1.0f / sum;
        for (int j = 0; j < NKEY; ++j) row[(j + tid) & (NKEY - 1)] *= inv;
    }
    __syncthreads();

    // --- GEMM2: out = P @ vg  (M=tokens, K=keys(256), N=hd(32)) ---
    // A tile loaded once per K step, feeds both hd B-tiles (2 WMMA per A load).
    v8f acc0 = {}, acc1 = {};
#pragma unroll
    for (int k0 = 0; k0 < NKEY; k0 += 32) {
        const float* prow = S + (size_t)(m0 + r) * NKEY + k0;
        v16h ap;
#pragma unroll
        for (int i = 0; i < 8; ++i) ap[i] = (_Float16)prow[half * 16 + i];
#pragma unroll
        for (int i = 0; i < 8; ++i) ap[8 + i] = (_Float16)prow[half * 16 + 8 + i];

        int reg2 = regs[k0 >> 6];
        const _Float16* vb = vT16 + (hb + reg2) * (RT * HD);
        const _Float16* pb0 = vb + (size_t)r * RT + (k0 & 63) + half * 16;
        const _Float16* pb1 = vb + (size_t)(16 + r) * RT + (k0 & 63) + half * 16;
        v16h bv0, bv1;
#pragma unroll
        for (int i = 0; i < 16; ++i) bv0[i] = pb0[i];
#pragma unroll
        for (int i = 0; i < 16; ++i) bv1[i] = pb1[i];
        acc0 = wmma_f16(ap, bv0, acc0);
        acc1 = wmma_f16(ap, bv1, acc1);
    }
    int wy = i_reg / NWIN, wx = i_reg % NWIN;
#pragma unroll
    for (int hd0 = 0; hd0 < HD; hd0 += 16) {
        int c_chan = h * HD + hd0 + r;   // D column = hd
#pragma unroll
        for (int j = 0; j < 8; ++j) {
            int tok = m0 + half * 8 + j;
            int y = wy * 8 + (tok >> 3);
            int x = wx * 8 + (tok & 7);
            float val = (hd0 == 0) ? acc0[j] : acc1[j];
            attn32[((size_t)n * CC + c_chan) * HWP + y * WW + x] = val;
        }
    }
}

// ---------------------------------------------------------------------------
// LePE depthwise 3x3 conv + bias + attention output -> transposed f16 [n][p][c]
// ---------------------------------------------------------------------------
__global__ void lepe_kernel(const float* __restrict__ attn32, const float* __restrict__ v32,
                            const float* __restrict__ lw, const float* __restrict__ lb,
                            _Float16* __restrict__ totT) {
    size_t t = (size_t)blockIdx.x * blockDim.x + threadIdx.x;
    if (t >= (size_t)NB * HWP * CC) return;
    int c = (int)(t & (CC - 1));
    size_t rest = t >> 8;
    int p = (int)(rest % HWP);
    int n = (int)(rest / HWP);
    int y = p / WW, x = p % WW;
    const float* vb = v32 + ((size_t)n * CC + c) * HWP;
    const float* w = lw + c * 9;
    float s = lb[c] + attn32[((size_t)n * CC + c) * HWP + p];
#pragma unroll
    for (int ky = 0; ky < 3; ++ky) {
        int yy = y + ky - 1;
        if (yy < 0 || yy >= HH) continue;
#pragma unroll
        for (int kx = 0; kx < 3; ++kx) {
            int xx = x + kx - 1;
            if (xx < 0 || xx >= WW) continue;
            s += w[ky * 3 + kx] * vb[yy * WW + xx];
        }
    }
    totT[t] = (_Float16)s;
}

// ---------------------------------------------------------------------------
// Output projection GEMM: y[o,p] = sum_c out_w[o,c]*tot[n,c,p] + out_b[o]
// 2x2 register blocking, grid (98, 8, 8).
// ---------------------------------------------------------------------------
__global__ __launch_bounds__(32) void gemm_out_kernel(
    const _Float16* __restrict__ w16,   // [256][256]
    const _Float16* __restrict__ totT,  // [n][3136][256]
    const float*    __restrict__ bias,  // [256]
    float*          __restrict__ out)   // [n][256][3136]
{
    int lane = threadIdx.x & 31;
    int n = blockIdx.z, m0 = blockIdx.y * 32, p0 = blockIdx.x * 32;
    const _Float16* B = totT + (size_t)n * HWP * CC;

    v8f acc00 = {}, acc01 = {}, acc10 = {}, acc11 = {};
#pragma unroll
    for (int k0 = 0; k0 < CC; k0 += 32) {
        v16h a0 = load_a16(w16, CC, m0, k0, lane);
        v16h a1 = load_a16(w16, CC, m0 + 16, k0, lane);
        v16h b0 = load_bT16(B, CC, k0, p0, lane);
        v16h b1 = load_bT16(B, CC, k0, p0 + 16, lane);
        acc00 = wmma_f16(a0, b0, acc00);
        acc01 = wmma_f16(a0, b1, acc01);
        acc10 = wmma_f16(a1, b0, acc10);
        acc11 = wmma_f16(a1, b1, acc11);
    }
    int half = lane >> 4, c = lane & 15;
#pragma unroll
    for (int mt = 0; mt < 2; ++mt) {
#pragma unroll
        for (int j = 0; j < 8; ++j) {
            int o = m0 + mt * 16 + half * 8 + j;
            float b = bias[o];
            float v0 = (mt == 0) ? acc00[j] : acc10[j];
            float v1 = (mt == 0) ? acc01[j] : acc11[j];
            out[((size_t)n * CC + o) * HWP + p0 + c] = v0 + b;
            out[((size_t)n * CC + o) * HWP + p0 + 16 + c] = v1 + b;
        }
    }
}

// ---------------------------------------------------------------------------
extern "C" void kernel_launch(void* const* d_in, const int* in_sizes, int n_in,
                              void* d_out, int out_size, void* d_ws, size_t ws_size,
                              hipStream_t stream) {
    (void)in_sizes; (void)n_in; (void)out_size; (void)ws_size;
    const float* x      = (const float*)d_in[0];
    const float* qkv_w  = (const float*)d_in[1];
    const float* qkv_b  = (const float*)d_in[2];
    const float* out_w  = (const float*)d_in[3];
    const float* out_b  = (const float*)d_in[4];
    const float* lepe_w = (const float*)d_in[5];
    const float* lepe_b = (const float*)d_in[6];
    float* out = (float*)d_out;

    constexpr size_t NE = (size_t)NB * HWP * CC;  // 6,422,528 elements

    char* p = (char*)d_ws;
    auto take = [&](size_t bytes) -> void* {
        void* r = (void*)p;
        p += (bytes + 255) & ~(size_t)255;
        return r;
    };
    _Float16* xT16   = (_Float16*)take(NE * 2);
    _Float16* wq16   = (_Float16*)take((size_t)3 * CC * CC * 2);
    _Float16* wo16   = (_Float16*)take((size_t)CC * CC * 2);
    _Float16* q16    = (_Float16*)take(NE * 2);
    _Float16* k16    = (_Float16*)take(NE * 2);
    _Float16* vT16   = (_Float16*)take(NE * 2);
    float*    v32    = (float*)take(NE * 4);
    float*    attn32 = (float*)take(NE * 4);
    _Float16* totT16 = (_Float16*)take(NE * 2);
    float*    q_r    = (float*)take((size_t)NB * NR * CC * 4);
    float*    k_r    = (float*)take((size_t)NB * NR * CC * 4);
    int*      idxb   = (int*)take((size_t)NB * NR * TOPK * 4);

    pack_x_kernel<<<dim3((unsigned)((NE + 255) / 256)), dim3(256), 0, stream>>>(x, xT16);
    cvt_f16_kernel<<<dim3((3 * CC * CC + 255) / 256), dim3(256), 0, stream>>>(qkv_w, wq16,
                                                                              3 * CC * CC);
    cvt_f16_kernel<<<dim3((CC * CC + 255) / 256), dim3(256), 0, stream>>>(out_w, wo16,
                                                                          CC * CC);
    gemm_qkv_kernel<<<dim3(HWP / 32, (3 * CC) / 32, NB), dim3(32), 0, stream>>>(
        wq16, xT16, qkv_b, q16, k16, vT16, v32);
    pool_kernel<<<dim3((NB * NR * CC + 255) / 256), dim3(256), 0, stream>>>(q16, k16, q_r,
                                                                            k_r);
    route_kernel<<<dim3((NB * NR + 255) / 256), dim3(256), 0, stream>>>(q_r, k_r, idxb);
    attn_kernel<<<dim3(NR, NH, NB), dim3(128), 0, stream>>>(q16, k16, vT16, idxb, attn32);
    lepe_kernel<<<dim3((unsigned)((NE + 255) / 256)), dim3(256), 0, stream>>>(
        attn32, v32, lepe_w, lepe_b, totT16);
    gemm_out_kernel<<<dim3(HWP / 32, CC / 32, NB), dim3(32), 0, stream>>>(wo16, totT16,
                                                                          out_b, out);
}
